// EfficientCrossAttention_14173392076937
// MI455X (gfx1250) — matrix operand order repaired and checked by reading, not compiled, not run on hardware
//
#include <hip/hip_runtime.h>
#include <hip/hip_bf16.h>

typedef __attribute__((ext_vector_type(16))) _Float16 v16h;
typedef __attribute__((ext_vector_type(8)))  _Float16 v8h;
typedef __attribute__((ext_vector_type(8)))  float    v8f;
typedef __attribute__((ext_vector_type(4)))  int      vi4;

#define CCH 512
#define LHW 4096
#define NB  8
#define NH  8   // heads
#define NCHUNK 8 // l-chunks for context reduction

// Async global->LDS path (gfx1250). Guarded: falls back to ds_store path.
#if defined(__has_builtin)
#  if __has_builtin(__builtin_amdgcn_global_load_async_to_lds_b128) && \
      __has_builtin(__builtin_amdgcn_s_wait_asynccnt)
#    define HAVE_ASYNC_LDS 1
#  endif
#endif
#ifndef HAVE_ASYNC_LDS
#  define HAVE_ASYNC_LDS 0
#endif

#if HAVE_ASYNC_LDS
typedef __attribute__((address_space(1))) vi4* as1_vi4_ptr;
typedef __attribute__((address_space(3))) vi4* as3_vi4_ptr;
__device__ inline void async_b128(const _Float16* g, _Float16* l) {
  __builtin_amdgcn_global_load_async_to_lds_b128(
      (as1_vi4_ptr)g, (as3_vi4_ptr)l, 0, 0);
}
#endif

// ---------------------------------------------------------------------------
// WMMA fragment loaders from LDS.
// A (16x32 f16, M x K): lanes 0-15 -> M=lane, K pattern {0..7, 16..23};
//                       lanes 16-31 -> M=lane-16, K pattern {8..15, 24..31}.
// LDS layout: A[m][k], row stride = ld halfs.  -> two contiguous 8-half loads.
// ---------------------------------------------------------------------------
__device__ inline v16h lds_load_A(const _Float16* tile, int ld) {
  int lane = threadIdx.x & 31;
  int m  = lane & 15;
  int h8 = (lane >> 4) * 8;           // 0 or 8
  union { v16h v; v8h p[2]; } u;
  u.p[0] = *(const v8h*)(tile + m * ld + h8);
  u.p[1] = *(const v8h*)(tile + m * ld + 16 + h8);
  return u.v;
}

// B (32x16 f16, K x N): lanes 0-15 -> N=lane, K {0..7, 16..23}; lanes 16-31 -> K {8..15, 24..31}.
// LDS layout: Bt[n][k] (transposed, K contiguous), row stride = ld halfs.
__device__ inline v16h lds_load_B(const _Float16* tile, int ld) {
  int lane = threadIdx.x & 31;
  int n   = lane & 15;
  int k16 = (lane >> 4) * 8;          // 0 or 8
  union { v16h v; v8h p[2]; } u;
  u.p[0] = *(const v8h*)(tile + n * ld + k16);
  u.p[1] = *(const v8h*)(tile + n * ld + k16 + 16);
  return u.v;
}

__device__ inline v8f wmma_f16(v16h a, v16h b, v8f c) {
  return __builtin_amdgcn_wmma_f32_16x16x32_f16(false, a, false, b, (short)0, c,
                                                false, false);
}

// ---------------------------------------------------------------------------
// Generic tiled GEMM:  Out[z][m][l] = sum_k W[m][k] * In[z][k][l] + bias[m]
// W: f32 [512][512] row-major.  In: [z][512][4096].  Out: [z][512][4096].
// Block = 256 threads (8 waves); block tile 64(M) x 128(N); K step 32;
// LDS double-buffered.  Wave w: m_off=(w&3)*16, n_off=(w>>2)*64 -> 4 WMMA/step.
// ---------------------------------------------------------------------------
template <typename TIN, typename TOUT>
__global__ void gemm_wmma(const float* __restrict__ W, const float* __restrict__ bias,
                          const TIN* __restrict__ In, TOUT* __restrict__ Out) {
  __shared__ __align__(16) _Float16 Af[2][64 * 32];    // [m][k]
  __shared__ __align__(16) _Float16 Bf[2][128 * 32];   // [n][k] transposed

  const int m0 = blockIdx.x * 64;
  const int l0 = blockIdx.y * 128;
  const TIN*  Inb  = In  + (size_t)blockIdx.z * CCH * LHW;
  TOUT*       Outb = Out + (size_t)blockIdx.z * CCH * LHW;

  const int t = threadIdx.x;
  const int w = t >> 5;
  const int m_off = (w & 3) * 16;
  const int n_off = (w >> 2) * 64;

  const v8f vzero = {0.f,0.f,0.f,0.f,0.f,0.f,0.f,0.f};
  v8f acc[4];
#pragma unroll
  for (int i = 0; i < 4; ++i) acc[i] = vzero;

  const int a_row = t >> 2, a_cb = (t & 3) * 8;    // A: 64 rows x 32 cols
  const int b_k   = t >> 3, b_lb = (t & 7) * 16;   // B: 32 k    x 128 l

  float ar[8];
  TIN   br[16];

  auto loadA = [&](int k0) {
#pragma unroll
    for (int j = 0; j < 8; ++j)
      ar[j] = W[(size_t)(m0 + a_row) * CCH + k0 + a_cb + j];
  };
  auto loadB = [&](int k0) {
#pragma unroll
    for (int j = 0; j < 16; ++j)
      br[j] = Inb[(size_t)(k0 + b_k) * LHW + l0 + b_lb + j];
  };
  auto stageA = [&](int buf) {
#pragma unroll
    for (int j = 0; j < 8; ++j)
      Af[buf][a_row * 32 + a_cb + j] = (_Float16)ar[j];
  };
  auto stageB = [&](int buf) {
#pragma unroll
    for (int j = 0; j < 16; ++j)
      Bf[buf][(b_lb + j) * 32 + b_k] = (_Float16)br[j];
  };

  loadA(0); loadB(0);
  stageA(0); stageB(0);
  __syncthreads();

  for (int k0 = 0; k0 < CCH; k0 += 32) {
    const int  cur  = (k0 >> 5) & 1;
    const int  nxt  = cur ^ 1;
    const bool more = (k0 + 32 < CCH);
    if (more) { loadA(k0 + 32); loadB(k0 + 32); }   // global -> regs (overlaps WMMA)

    v16h a = lds_load_A(Af[cur] + m_off * 32, 32);
#pragma unroll
    for (int i = 0; i < 4; ++i) {
      v16h b = lds_load_B(Bf[cur] + (n_off + 16 * i) * 32, 32);
      acc[i] = wmma_f16(a, b, acc[i]);
    }

    if (more) { stageA(nxt); stageB(nxt); }         // regs -> LDS (other buffer)
    __syncthreads();
  }

  const int lane = t & 31, nn = lane & 15, half = lane >> 4;
#pragma unroll
  for (int r = 0; r < 8; ++r) {
    int m = m0 + m_off + r + 8 * half;
    float bb = bias[m];
#pragma unroll
    for (int i = 0; i < 4; ++i)
      Outb[(size_t)m * LHW + l0 + n_off + 16 * i + nn] = (TOUT)(acc[i][r] + bb);
  }
}

// ---------------------------------------------------------------------------
// Softmax over spatial axis (keys): one block per (n,c) row of 4096, in-place.
// ---------------------------------------------------------------------------
__global__ void softmax_spatial(_Float16* kb) {
  __shared__ float red[256];
  _Float16* p = kb + (size_t)blockIdx.x * LHW;
  const int t = threadIdx.x;

  float mx = -1e30f;
  for (int i = t; i < LHW; i += 256) mx = fmaxf(mx, (float)p[i]);
  red[t] = mx; __syncthreads();
  for (int s = 128; s > 0; s >>= 1) {
    if (t < s) red[t] = fmaxf(red[t], red[t + s]);
    __syncthreads();
  }
  mx = red[0]; __syncthreads();

  float sum = 0.f;
  for (int i = t; i < LHW; i += 256) sum += __expf((float)p[i] - mx);
  red[t] = sum; __syncthreads();
  for (int s = 128; s > 0; s >>= 1) {
    if (t < s) red[t] += red[t + s];
    __syncthreads();
  }
  float inv = 1.0f / red[0];

  for (int i = t; i < LHW; i += 256)
    p[i] = (_Float16)(__expf((float)p[i] - mx) * inv);
}

// ---------------------------------------------------------------------------
// Softmax over head-channel axis (queries): one thread per (n,h,l), 64 deep,
// stride 4096 halfs, in-place.
// ---------------------------------------------------------------------------
__global__ void softmax_channel(_Float16* qb) {
  int id = blockIdx.x * blockDim.x + threadIdx.x;   // 8*8*4096 total
  int n  = id >> 15;
  int r  = id & 32767;
  int hh = r >> 12;
  int l  = r & 4095;
  _Float16* p = qb + ((size_t)n * CCH + hh * 64) * LHW + l;

  float mx = -1e30f;
  for (int c = 0; c < 64; ++c) mx = fmaxf(mx, (float)p[(size_t)c * LHW]);
  float sum = 0.f;
  for (int c = 0; c < 64; ++c) sum += __expf((float)p[(size_t)c * LHW] - mx);
  float inv = 1.0f / sum;
  for (int c = 0; c < 64; ++c)
    p[(size_t)c * LHW] = (_Float16)(__expf((float)p[(size_t)c * LHW] - mx) * inv);
}

// ---------------------------------------------------------------------------
// Partial context: ctxp[chunk][n,h] (64x64) = k_sm[n,h][:, chunk] @ v^T chunk.
// Grid (n*h, chunk); each block reduces l in [chunk*512, chunk*512+512).
// v's [c][l] storage IS the transposed [n][k] B layout.
// Uses gfx1250 async global->LDS copies when available.
// ---------------------------------------------------------------------------
__global__ void gemm_context(const _Float16* __restrict__ kb,
                             const _Float16* __restrict__ vb,
                             float* __restrict__ ctxp) {
  __shared__ __align__(16) _Float16 Ak[64 * 32];
  __shared__ __align__(16) _Float16 Bv[64 * 32];

  const int nh    = blockIdx.x;  // n*8 + h
  const int chunk = blockIdx.y;
  const _Float16* kp = kb + (size_t)nh * 64 * LHW;
  const _Float16* vp = vb + (size_t)nh * 64 * LHW;

  const int t = threadIdx.x, w = t >> 5;
  const int m_off = (w & 3) * 16, n_off = (w >> 2) * 32;
  const int c = t >> 2, jb = (t & 3) * 8;   // 64 rows x 32 l

  const v8f vzero = {0.f,0.f,0.f,0.f,0.f,0.f,0.f,0.f};
  v8f acc0 = vzero, acc1 = vzero;

  for (int it = 0; it < 16; ++it) {
    const int l0 = chunk * (LHW / NCHUNK) + it * 32;
#if HAVE_ASYNC_LDS
    async_b128(kp + (size_t)c * LHW + l0 + jb, Ak + c * 32 + jb);
    async_b128(vp + (size_t)c * LHW + l0 + jb, Bv + c * 32 + jb);
    __builtin_amdgcn_s_wait_asynccnt(0);
#else
    *(v8h*)(Ak + c * 32 + jb) = *(const v8h*)(kp + (size_t)c * LHW + l0 + jb);
    *(v8h*)(Bv + c * 32 + jb) = *(const v8h*)(vp + (size_t)c * LHW + l0 + jb);
#endif
    __syncthreads();
    v16h a  = lds_load_A(Ak + m_off * 32, 32);
    v16h b0 = lds_load_B(Bv + n_off * 32, 32);
    v16h b1 = lds_load_B(Bv + (n_off + 16) * 32, 32);
    acc0 = wmma_f16(a, b0, acc0);
    acc1 = wmma_f16(a, b1, acc1);
    __syncthreads();
  }

  const int lane = t & 31, nn = lane & 15, half = lane >> 4;
  float* cp = ctxp + ((size_t)chunk * (NB * NH) + nh) * 64 * 64;
#pragma unroll
  for (int r = 0; r < 8; ++r) {
    int m = m_off + r + 8 * half;
    cp[m * 64 + n_off + nn]      = acc0[r];
    cp[m * 64 + n_off + 16 + nn] = acc1[r];
  }
}

// ---------------------------------------------------------------------------
// att[n,h] (64 x 4096) = context[n,h]^T (64x64) @ q_sm[n,h] (64 x 4096).
// A tile = deterministic sum of the 8 per-chunk partial contexts (transposed).
// Block per (l-tile, n*h). M=val-ch, N=l, K=key-ch (K=64 -> 2 WMMA steps).
// ---------------------------------------------------------------------------
__global__ void gemm_attend(const float* __restrict__ ctxp,
                            const _Float16* __restrict__ qb,
                            _Float16* __restrict__ att) {
  __shared__ __align__(16) _Float16 Act[64 * 64];  // [m=vch][k=kch]
  __shared__ __align__(16) _Float16 Bq[64 * 64];   // [n=l][k=kch]

  const int l0 = blockIdx.x * 64;
  const int nh = blockIdx.y;
  const _Float16* qp = qb  + (size_t)nh * 64 * LHW;
  _Float16*       ap = att + (size_t)nh * 64 * LHW;

  const int t = threadIdx.x;
  {
    int k = t >> 2, base = (t & 3) * 16;
#pragma unroll
    for (int j = 0; j < 16; ++j) {
      float s = 0.f;
#pragma unroll
      for (int ch = 0; ch < NCHUNK; ++ch)
        s += ctxp[((size_t)ch * (NB * NH) + nh) * 4096 + k * 64 + base + j];
      Act[(base + j) * 64 + k] = (_Float16)s;   // transpose: ctx[k][m] -> A[m][k]
    }
#pragma unroll
    for (int j = 0; j < 16; ++j)
      Bq[(base + j) * 64 + k] = qp[(size_t)k * LHW + l0 + base + j];
  }
  __syncthreads();

  const int w = t >> 5;
  const int m_off = (w & 3) * 16, n_off = (w >> 2) * 32;
  const v8f vzero = {0.f,0.f,0.f,0.f,0.f,0.f,0.f,0.f};
  v8f acc0 = vzero, acc1 = vzero;

#pragma unroll
  for (int kbk = 0; kbk < 64; kbk += 32) {
    v16h a  = lds_load_A(Act + m_off * 64 + kbk, 64);
    v16h b0 = lds_load_B(Bq + n_off * 64 + kbk, 64);
    v16h b1 = lds_load_B(Bq + (n_off + 16) * 64 + kbk, 64);
    acc0 = wmma_f16(a, b0, acc0);
    acc1 = wmma_f16(a, b1, acc1);
  }

  const int lane = t & 31, nn = lane & 15, half = lane >> 4;
#pragma unroll
  for (int r = 0; r < 8; ++r) {
    int m = m_off + r + 8 * half;   // value channel
    ap[(size_t)m * LHW + l0 + n_off + nn]      = (_Float16)acc0[r];
    ap[(size_t)m * LHW + l0 + n_off + 16 + nn] = (_Float16)acc1[r];
  }
}

// ---------------------------------------------------------------------------
extern "C" void kernel_launch(void* const* d_in, const int* in_sizes, int n_in,
                              void* d_out, int out_size, void* d_ws, size_t ws_size,
                              hipStream_t stream) {
  const float* x  = (const float*)d_in[0];
  const float* y  = (const float*)d_in[1];
  const float* Wk = (const float*)d_in[2];
  const float* bk = (const float*)d_in[3];
  const float* Wq = (const float*)d_in[4];
  const float* bq = (const float*)d_in[5];
  const float* Wv = (const float*)d_in[6];
  const float* bv = (const float*)d_in[7];
  const float* Wr = (const float*)d_in[8];
  const float* br = (const float*)d_in[9];
  float* out = (float*)d_out;

  const size_t half_buf = (size_t)NB * CCH * LHW * sizeof(_Float16);           // 32 MB
  const size_t ctx_buf  = (size_t)NCHUNK * NB * NH * 64 * 64 * sizeof(float);  // 8 MB
  char* ws = (char*)d_ws;
  _Float16* kbuf = (_Float16*)(ws);
  _Float16* qbuf = (_Float16*)(ws + half_buf);
  _Float16* vbuf = (_Float16*)(ws + 2 * half_buf);
  float*    ctxp = (float*)   (ws + 3 * half_buf);
  _Float16* attb = (_Float16*)(ws + 3 * half_buf + ctx_buf);

  dim3 blk(256);
  dim3 gproj(CCH / 64, LHW / 128, NB);   // (8, 32, 8)

  // Q/K/V projections (f32 in -> f16 out), WMMA f16.
  gemm_wmma<float, _Float16><<<gproj, blk, 0, stream>>>(Wk, bk, y, kbuf);
  gemm_wmma<float, _Float16><<<gproj, blk, 0, stream>>>(Wq, bq, x, qbuf);
  gemm_wmma<float, _Float16><<<gproj, blk, 0, stream>>>(Wv, bv, y, vbuf);

  // Softmaxes (in-place).
  softmax_spatial<<<dim3(NB * CCH), blk, 0, stream>>>(kbuf);
  softmax_channel<<<dim3((NB * NH * LHW) / 256), blk, 0, stream>>>(qbuf);

  // context partials over 8 l-chunks, then att = context^T @ q.
  gemm_context<<<dim3(NB * NH, NCHUNK), blk, 0, stream>>>(kbuf, vbuf, ctxp);
  gemm_attend<<<dim3(LHW / 64, NB * NH), blk, 0, stream>>>(ctxp, qbuf, attb);

  // Reprojection (f16 in -> f32 out).
  gemm_wmma<_Float16, float><<<gproj, blk, 0, stream>>>(Wr, br, attb, out);
}
